// SimpleGNN_63239098466369
// MI455X (gfx1250) — matrix operand order repaired and checked by reading, not compiled, not run on hardware
//
#include <hip/hip_runtime.h>
#include <hip/hip_bf16.h>

typedef __attribute__((ext_vector_type(16))) __bf16 v16bf;
typedef __attribute__((ext_vector_type(8)))  float  v8f;

struct alignas(16) U4 { unsigned int x, y, z, w; };
union FragU { v16bf v; U4 q[2]; };

#define GBM 128   // rows per block (8 waves x 16)
#define GBN 64    // cols per block (4 WMMA tiles per wave)
#define GBK 32    // K per step (one WMMA K)
#define LDA 40    // A LDS row stride in bf16 elems (80B, 16B-aligned rows)
#define LDB 40    // B LDS row stride in bf16 elems

// -------- weight prepack: Wp[(kb*N + n)*32 + kk] = bf16(W[kb*32+kk][n]) --------
// Each column n holds its 32 K-values contiguously; a lane reads two 8-value runs (half*8, 16+half*8).
__global__ void k_pack_w(const float* __restrict__ W, __bf16* __restrict__ Wp,
                         int N, int total) {
  int idx = blockIdx.x * 256 + threadIdx.x;
  if (idx >= total) return;
  int kk  = idx & 31;
  int rem = idx >> 5;          // kb*N + n
  int n   = rem % N;
  int kb  = rem / N;
  Wp[idx] = (__bf16)W[((size_t)kb * 32 + kk) * N + n];
}

// ---------------- bf16 WMMA GEMM: C[M,N] = A[M,K] @ W[K,N] (A fp32, W prepacked bf16) ----
__global__ __launch_bounds__(256)
void k_gemm_wmma(const float* __restrict__ A, const __bf16* __restrict__ Bp,
                 float* __restrict__ C, int M, int N, int K) {
  __shared__ __align__(16) __bf16 As[GBM * LDA];  // 10240 B
  __shared__ __align__(16) __bf16 Bs[GBN * LDB];  //  5120 B

  const int tid  = threadIdx.x;
  const int m0   = blockIdx.x * GBM;
  const int n0   = blockIdx.y * GBN;
  const int lane = tid & 31;
  const int wave = tid >> 5;          // one 16-row m-tile per wave
  const int half = lane >> 4;
  const int l15  = lane & 15;

  v8f acc[4] = {{}, {}, {}, {}};

  // A loader: 128 rows x 2 segments of 16 floats
  const int arow = tid >> 1, aseg = tid & 1;
  // B loader: 64 packed rows x 4 parts of 8 bf16 (16B each) -> 256 async copies of 16B
  const int brow = tid >> 2, bpart = tid & 3;

  const unsigned bsBase = (unsigned)(unsigned long long)(const void*)&Bs[0];
  const unsigned bsOff  = bsBase + (unsigned)(brow * LDB + bpart * 8) * 2;

  const int gr = m0 + arow;
  const int nsteps = K / GBK;

  for (int kb = 0; kb < nsteps; ++kb) {
    // ---- async copy B panel (already bf16, fragment-ordered, contiguous per row) ----
    {
      const __bf16* bsrc = Bp + ((size_t)kb * N + n0 + brow) * 32 + bpart * 8;
      asm volatile("global_load_async_to_lds_b128 %0, %1, off"
                   :: "v"(bsOff), "v"(bsrc) : "memory");
    }
    // ---- stage A tile (fp32 -> bf16), 16 elems/thread ----
    {
      __bf16* d = &As[arow * LDA + aseg * 16];
      if (gr < M) {
        const float* s = A + (size_t)gr * K + kb * GBK + aseg * 16;
        #pragma unroll
        for (int j = 0; j < 16; ++j) d[j] = (__bf16)s[j];
      } else {
        #pragma unroll
        for (int j = 0; j < 16; ++j) d[j] = (__bf16)0.0f;
      }
    }
    asm volatile("s_wait_asynccnt 0x0" ::: "memory");
    __syncthreads();

    // ---- fetch ALL fragments first (one DS clause, partial waits), then 4 WMMAs ----
    // A fragment: elems 0..7 -> K=half*8+e ; elems 8..15 -> K=16+half*8+(e-8)
    FragU fa;
    fa.q[0] = *(const U4*)&As[(wave * 16 + l15) * LDA + half * 8];
    fa.q[1] = *(const U4*)&As[(wave * 16 + l15) * LDA + 16 + half * 8];

    // B fragments: elems 0..7 -> K=half*8+e ; elems 8..15 -> K=16+half*8+(e-8)
    FragU fb[4];
    #pragma unroll
    for (int t = 0; t < 4; ++t) {
      fb[t].q[0] = *(const U4*)&Bs[(t * 16 + l15) * LDB + half * 8];
      fb[t].q[1] = *(const U4*)&Bs[(t * 16 + l15) * LDB + 16 + half * 8];
    }

    #pragma unroll
    for (int t = 0; t < 4; ++t)
      acc[t] = __builtin_amdgcn_wmma_f32_16x16x32_bf16(false, fa.v, false, fb[t].v,
                                                       (short)0, acc[t], false, false);
    __syncthreads();
  }

  // ---- store D: VGPR j -> row j + 8*half, l15 -> col ----
  #pragma unroll
  for (int j = 0; j < 8; ++j) {
    const int row = m0 + wave * 16 + j + 8 * half;
    if (row < M) {
      #pragma unroll
      for (int t = 0; t < 4; ++t)
        C[(size_t)row * N + n0 + t * 16 + l15] = acc[t][j];
    }
  }
}

// ---------------- degree / normalization ----------------
__global__ void k_init_deg(float* deg, int n) {
  int i = blockIdx.x * 256 + threadIdx.x;
  if (i < n) deg[i] = 1.0f;             // self-loop contributes 1
}
__global__ void k_edge_deg(const int* __restrict__ dst, float* __restrict__ deg, int e) {
  int i = blockIdx.x * 256 + threadIdx.x;
  if (i < e) atomicAdd(&deg[dst[i]], 1.0f);
}
__global__ void k_rsqrt(float* deg, int n) {
  int i = blockIdx.x * 256 + threadIdx.x;
  if (i < n) deg[i] = rsqrtf(deg[i]);   // deg >= 1 always (self-loops)
}

// ---------------- edge scatter: out[dst] += dinv[s]*dinv[d] * h[src] ----------------
__global__ void k_scatter(const float* __restrict__ h, const int* __restrict__ src,
                          const int* __restrict__ dst, const float* __restrict__ dinv,
                          float* __restrict__ out, int E, int shift) {
  const int D4 = 1 << shift;            // float4 chunks per row
  int tid = blockIdx.x * 256 + threadIdx.x;
  int e = tid >> shift;
  if (e >= E) return;
  int c = tid & (D4 - 1);
  int s = src[e], d = dst[e];
  float nrm = dinv[s] * dinv[d];
  float4 v = ((const float4*)h)[(size_t)s * D4 + c];
  float* op = out + ((size_t)d * D4 + c) * 4;
  atomicAdd(op + 0, v.x * nrm);
  atomicAdd(op + 1, v.y * nrm);
  atomicAdd(op + 2, v.z * nrm);
  atomicAdd(op + 3, v.w * nrm);
}

// ---------------- self-loop + bias + ReLU (in place on agg) ----------------
__global__ void k_finalize(float* __restrict__ agg, const float* __restrict__ h,
                           const float* __restrict__ dinv, const float* __restrict__ bias,
                           int n, int shift) {
  const int D4 = 1 << shift;
  int tid = blockIdx.x * 256 + threadIdx.x;
  int i = tid >> shift;
  if (i >= n) return;
  int c = tid & (D4 - 1);
  float di = dinv[i];
  float w = di * di;                     // self-loop norm = dinv[i]^2
  float4 a  = ((float4*)agg)[(size_t)i * D4 + c];
  float4 hv = ((const float4*)h)[(size_t)i * D4 + c];
  float4 bv = ((const float4*)bias)[c];
  float4 r;
  r.x = fmaxf(a.x + w * hv.x + bv.x, 0.0f);
  r.y = fmaxf(a.y + w * hv.y + bv.y, 0.0f);
  r.z = fmaxf(a.z + w * hv.z + bv.z, 0.0f);
  r.w = fmaxf(a.w + w * hv.w + bv.w, 0.0f);
  ((float4*)agg)[(size_t)i * D4 + c] = r;
}

// ---------------- segment sums for mean pool (D = 256 fixed) ----------------
__global__ void k_pool(const float* __restrict__ h, const int* __restrict__ batch,
                       float* __restrict__ pool, float* __restrict__ cnt, int n) {
  int tid = blockIdx.x * 256 + threadIdx.x;
  int i = tid >> 6;                      // 64 float4 chunks per row
  if (i >= n) return;
  int c = tid & 63;
  int g = batch[i];
  float4 v = ((const float4*)h)[(size_t)i * 64 + c];
  float* p = pool + (size_t)g * 256 + c * 4;
  atomicAdd(p + 0, v.x);
  atomicAdd(p + 1, v.y);
  atomicAdd(p + 2, v.z);
  atomicAdd(p + 3, v.w);
  if (c == 0) atomicAdd(&cnt[g], 1.0f);
}

// ---------------- classifier: out[64,8] = (pool/cnt) @ Wl + bl ----------------
__global__ void k_classify(const float* __restrict__ pool, const float* __restrict__ cnt,
                           const float* __restrict__ Wl, const float* __restrict__ bl,
                           float* __restrict__ out) {
  int tid = threadIdx.x;                 // 512 threads: 64 graphs x 8 classes
  int g = tid >> 3, j = tid & 7;
  float s = 0.0f;
  #pragma unroll 4
  for (int c = 0; c < 256; ++c) s += pool[g * 256 + c] * Wl[c * 8 + j];
  out[g * 8 + j] = s / fmaxf(cnt[g], 1.0f) + bl[j];
}

extern "C" void kernel_launch(void* const* d_in, const int* in_sizes, int n_in,
                              void* d_out, int out_size, void* d_ws, size_t ws_size,
                              hipStream_t stream) {
  const float* x  = (const float*)d_in[0];   // [N,512]
  const float* W1 = (const float*)d_in[1];   // [512,512]
  const float* b1 = (const float*)d_in[2];   // [512]
  const float* W2 = (const float*)d_in[3];   // [512,256]
  const float* b2 = (const float*)d_in[4];   // [256]
  const float* Wl = (const float*)d_in[5];   // [256,8]
  const float* bl = (const float*)d_in[6];   // [8]
  const int* eidx = (const int*)d_in[7];     // [2,E] flat
  const int* batch= (const int*)d_in[8];     // [N]

  const int IN_DIM = 512, HID = 512, OUT = 256;
  const int NN = in_sizes[0] / IN_DIM;       // 10000
  const int E  = in_sizes[7] / 2;            // 160000
  const int* src = eidx;
  const int* dst = eidx + E;

  // workspace layout (float units; all offsets 1KB-aligned)
  float* ws = (float*)d_ws;
  size_t off = 0;
  float* deg  = ws + off; off += (size_t)((NN + 255) & ~255);
  float* h1   = ws + off; off += (size_t)NN * HID;
  float* agg1 = ws + off; off += (size_t)NN * HID;
  float* h2   = ws + off; off += (size_t)NN * OUT;
  float* agg2 = ws + off; off += (size_t)NN * OUT;
  float* pool = ws + off; off += 64 * 256;
  float* cnt  = ws + off; off += 256;
  __bf16* Wp1 = (__bf16*)(ws + off); off += (size_t)IN_DIM * HID / 2;  // 512KB bf16
  __bf16* Wp2 = (__bf16*)(ws + off); off += (size_t)HID * OUT / 2;     // 256KB bf16

  // zero accumulation buffers (graph-capturable memsets)
  hipMemsetAsync(agg1, 0, (size_t)NN * HID * sizeof(float), stream);
  hipMemsetAsync(agg2, 0, (size_t)NN * OUT * sizeof(float), stream);
  hipMemsetAsync(pool, 0, (64 * 256 + 256) * sizeof(float), stream);  // pool + cnt

  // prepack weights to bf16 fragment layout (once per launch; L2-resident thereafter)
  k_pack_w<<<(IN_DIM * HID + 255) / 256, 256, 0, stream>>>(W1, Wp1, HID, IN_DIM * HID);
  k_pack_w<<<(HID * OUT + 255) / 256, 256, 0, stream>>>(W2, Wp2, OUT, HID * OUT);

  // degrees -> dinv (stored in deg)
  k_init_deg<<<(NN + 255) / 256, 256, 0, stream>>>(deg, NN);
  k_edge_deg<<<(E + 255) / 256, 256, 0, stream>>>(dst, deg, E);
  k_rsqrt<<<(NN + 255) / 256, 256, 0, stream>>>(deg, NN);

  // layer 1: h1 = x @ W1  (WMMA bf16)
  {
    dim3 g((NN + GBM - 1) / GBM, HID / GBN);
    k_gemm_wmma<<<g, 256, 0, stream>>>(x, Wp1, h1, NN, HID, IN_DIM);
  }
  // layer 1 aggregation + self-loop + bias + ReLU  (D=512 -> shift 7)
  {
    long long tot = (long long)E * (HID / 4);
    k_scatter<<<(unsigned)((tot + 255) / 256), 256, 0, stream>>>(h1, src, dst, deg, agg1, E, 7);
    long long tot2 = (long long)NN * (HID / 4);
    k_finalize<<<(unsigned)((tot2 + 255) / 256), 256, 0, stream>>>(agg1, h1, deg, b1, NN, 7);
  }

  // layer 2: h2 = agg1 @ W2
  {
    dim3 g((NN + GBM - 1) / GBM, OUT / GBN);
    k_gemm_wmma<<<g, 256, 0, stream>>>(agg1, Wp2, h2, NN, OUT, HID);
  }
  // layer 2 aggregation + self-loop + bias + ReLU  (D=256 -> shift 6)
  {
    long long tot = (long long)E * (OUT / 4);
    k_scatter<<<(unsigned)((tot + 255) / 256), 256, 0, stream>>>(h2, src, dst, deg, agg2, E, 6);
    long long tot2 = (long long)NN * (OUT / 4);
    k_finalize<<<(unsigned)((tot2 + 255) / 256), 256, 0, stream>>>(agg2, h2, deg, b2, NN, 6);
  }

  // global mean pool + classifier
  {
    long long tot = (long long)NN * (OUT / 4);
    k_pool<<<(unsigned)((tot + 255) / 256), 256, 0, stream>>>(agg2, batch, pool, cnt, NN);
    k_classify<<<1, 512, 0, stream>>>(pool, cnt, Wl, bl, (float*)d_out);
  }
}
